// MambaBlock_46892452937750
// MI455X (gfx1250) — compile-verified, matrix operand order checked by reading, not run on hardware
//
#include <hip/hip_runtime.h>
#include <cstddef>

// ---------------- CDNA5 WMMA types ----------------
typedef __bf16  v16bf __attribute__((ext_vector_type(16)));
typedef float   v8f   __attribute__((ext_vector_type(8)));

union FragU { v16bf v; uint4 q[2]; };

// Problem constants (from reference)
#define BATCH 2
#define SEQ   1024
#define HDIM  1024
#define INNER 2048
#define XZW   4096      // 2*INNER
#define NST   16        // SSM state
#define DTRD  64
#define TOK   (BATCH*SEQ)   // 2048 tokens

// ---------------- f32 -> bf16 convert ----------------
__global__ __launch_bounds__(256) void k_f32_to_bf16(const float* __restrict__ in,
                                                     __bf16* __restrict__ out, int n) {
    int i = blockIdx.x * 256 + threadIdx.x;
    if (i < n) out[i] = (__bf16)in[i];
}

// ---------------- LayerNorm (H=1024) -> bf16 ----------------
__global__ __launch_bounds__(256) void k_layernorm_bf16(const float* __restrict__ x,
                                                        const float* __restrict__ w,
                                                        const float* __restrict__ b,
                                                        __bf16* __restrict__ out) {
    __shared__ float red[256];
    const int t = blockIdx.x;                 // token 0..2047
    const float* xr = x + (size_t)t * HDIM;
    float v[4];
    float s = 0.f;
#pragma unroll
    for (int i = 0; i < 4; ++i) { v[i] = xr[threadIdx.x + 256 * i]; s += v[i]; }
    red[threadIdx.x] = s; __syncthreads();
    for (int o = 128; o > 0; o >>= 1) {
        if (threadIdx.x < o) red[threadIdx.x] += red[threadIdx.x + o];
        __syncthreads();
    }
    const float mu = red[0] * (1.f / HDIM);
    __syncthreads();
    s = 0.f;
#pragma unroll
    for (int i = 0; i < 4; ++i) { float d = v[i] - mu; s += d * d; }
    red[threadIdx.x] = s; __syncthreads();
    for (int o = 128; o > 0; o >>= 1) {
        if (threadIdx.x < o) red[threadIdx.x] += red[threadIdx.x + o];
        __syncthreads();
    }
    const float rstd = rsqrtf(red[0] * (1.f / HDIM) + 1e-5f);
#pragma unroll
    for (int i = 0; i < 4; ++i) {
        int c = threadIdx.x + 256 * i;
        out[(size_t)t * HDIM + c] = (__bf16)((v[i] - mu) * rstd * w[c] + b[c]);
    }
}

// ---------------- bf16 WMMA GEMM: C = A(MxK) @ B(KxN) + bias ----------------
// act: 0 = none, 1 = clip(softplus(v), 1e-3, 0.1)
// Block = 256 threads = 8 waves; block tile 64(M) x 128(N); wave tile 32x32
// (2x2 fragments -> 4 v_wmma per K-step). K mult of 32; M mult of 64; N mult of 16.
__global__ __launch_bounds__(256) void k_wmma_gemm(const __bf16* __restrict__ A,
                                                   const __bf16* __restrict__ Bm,
                                                   const float* __restrict__ bias,
                                                   float* __restrict__ C,
                                                   int M, int N, int K, int act) {
    __shared__ __bf16 sA[64 * 32];          // A tile, row-major [m][k]   (4 KB)
    __shared__ __bf16 sBt[128 * 32];        // B tile, transposed [n][k]  (8 KB)

    const int tid   = threadIdx.x;
    const int wave  = __builtin_amdgcn_readfirstlane(tid >> 5);  // scalar wave id
    const int lane  = tid & 31;
    const int half  = lane >> 4;            // K-group select (ISA 7.12.2 layout)
    const int r16   = lane & 15;

    const int waveM = wave >> 2;            // 0..1
    const int waveN = wave & 3;             // 0..3

    const int rowBase = blockIdx.y * 64;
    const int cbBlk   = blockIdx.x * 128;
    const int colW    = cbBlk + waveN * 32;

    // how many 16-wide column fragments of this wave are inside N (scalar)
    int rem = (N - colW) >> 4;
    const int nAct = __builtin_amdgcn_readfirstlane(rem < 0 ? 0 : (rem > 2 ? 2 : rem));

    v8f a00 = {0.f,0.f,0.f,0.f,0.f,0.f,0.f,0.f};
    v8f a01 = a00, a10 = a00, a11 = a00;

    // staging coordinates (loop-invariant)
    const int sa_idx = tid * 8;             // 8 bf16 per thread for A tile
    const int sa_m   = sa_idx >> 5;         // 0..63
    const int sa_j   = sa_idx & 31;         // 0,8,16,24
    const int bn0    = (tid & 63) * 2;      // column pair for B staging
    const int bk0    = (tid >> 6) * 8;      // 8 K rows per thread

    for (int kk = 0; kk < K; kk += 32) {
        // ---- stage A: one b128 global load + one b128 LDS store per thread
        {
            const uint4 va = *(const uint4*)(A + (size_t)(rowBase + sa_m) * K + kk + sa_j);
            *(uint4*)&sA[sa_m * 32 + sa_j] = va;
            if (kk + 32 < K)
                __builtin_prefetch(A + (size_t)(rowBase + sa_m) * K + kk + 32 + sa_j, 0, 1);
        }
        // ---- stage B transposed: 8 u32 global loads -> 2 b128 LDS stores
        if (cbBlk + bn0 < N) {
            unsigned int u[8];
#pragma unroll
            for (int i = 0; i < 8; ++i)
                u[i] = *(const unsigned int*)(Bm + (size_t)(kk + bk0 + i) * N + cbBlk + bn0);
            uint4 p0, p1;
            p0.x = (u[0] & 0xffffu) | (u[1] << 16);  p1.x = (u[0] >> 16) | (u[1] & 0xffff0000u);
            p0.y = (u[2] & 0xffffu) | (u[3] << 16);  p1.y = (u[2] >> 16) | (u[3] & 0xffff0000u);
            p0.z = (u[4] & 0xffffu) | (u[5] << 16);  p1.z = (u[4] >> 16) | (u[5] & 0xffff0000u);
            p0.w = (u[6] & 0xffffu) | (u[7] << 16);  p1.w = (u[6] >> 16) | (u[7] & 0xffff0000u);
            *(uint4*)&sBt[(bn0    ) * 32 + bk0] = p0;
            *(uint4*)&sBt[(bn0 + 1) * 32 + bk0] = p1;
            if (kk + 32 < K)
                __builtin_prefetch(Bm + (size_t)(kk + 32 + bk0) * N + cbBlk + bn0, 0, 1);
        }
        __syncthreads();

        if (nAct > 0) {
            // fragment gathers: v16bf elems 0..7 = K half*8..+7, 8..15 = K 16+half*8..+7
            FragU fa0, fa1, fb0, fb1;
            const int ar0 = (waveM * 32 + r16) * 32;
            const int ar1 = (waveM * 32 + 16 + r16) * 32;
            fa0.q[0] = *(const uint4*)&sA[ar0 + half * 8];
            fa0.q[1] = *(const uint4*)&sA[ar0 + 16 + half * 8];
            fa1.q[0] = *(const uint4*)&sA[ar1 + half * 8];
            fa1.q[1] = *(const uint4*)&sA[ar1 + 16 + half * 8];
            const int br0 = (waveN * 32 + r16) * 32;
            fb0.q[0] = *(const uint4*)&sBt[br0 + half * 8];
            fb0.q[1] = *(const uint4*)&sBt[br0 + 16 + half * 8];
            a00 = __builtin_amdgcn_wmma_f32_16x16x32_bf16(false, fa0.v, false, fb0.v, (short)0, a00, false, false);
            a10 = __builtin_amdgcn_wmma_f32_16x16x32_bf16(false, fa1.v, false, fb0.v, (short)0, a10, false, false);
            if (nAct > 1) {
                const int br1 = (waveN * 32 + 16 + r16) * 32;
                fb1.q[0] = *(const uint4*)&sBt[br1 + half * 8];
                fb1.q[1] = *(const uint4*)&sBt[br1 + 16 + half * 8];
                a01 = __builtin_amdgcn_wmma_f32_16x16x32_bf16(false, fa0.v, false, fb1.v, (short)0, a01, false, false);
                a11 = __builtin_amdgcn_wmma_f32_16x16x32_bf16(false, fa1.v, false, fb1.v, (short)0, a11, false, false);
            }
        }
        __syncthreads();
    }

    // ---- epilogue: C layout per ISA: lane half selects M+8, r16 = N col
    // NOTE: by-value lambda, no address taken of accumulators (avoids scratch spill)
    auto store_tile = [&](v8f acc, int mi, int ci) {
        const int n = colW + ci * 16 + r16;
        const float bv = bias ? bias[n] : 0.f;
#pragma unroll
        for (int r = 0; r < 8; ++r) {
            const int m = rowBase + waveM * 32 + mi * 16 + r + half * 8;
            float v = acc[r] + bv;
            if (act == 1) {
                float sp = (v > 20.f) ? v : log1pf(__expf(v));
                v = fminf(fmaxf(sp, 0.001f), 0.1f);
            }
            C[(size_t)m * N + n] = v;
        }
    };
    if (nAct > 0) {
        store_tile(a00, 0, 0);
        store_tile(a10, 1, 0);
        if (nAct > 1) {
            store_tile(a01, 0, 1);
            store_tile(a11, 1, 1);
        }
    }
}

// ---------------- causal depthwise conv (CONV=4) + SiLU ----------------
__global__ __launch_bounds__(256) void k_conv_silu(const float* __restrict__ xz,
                                                   const float* __restrict__ cw,
                                                   const float* __restrict__ cb,
                                                   float* __restrict__ xa,
                                                   __bf16* __restrict__ xab) {
    const int token = blockIdx.x;            // 0..2047
    const int b = token >> 10, t = token & 1023;
    for (int d = threadIdx.x; d < INNER; d += 256) {
        float accv = cb[d];
#pragma unroll
        for (int k = 0; k < 4; ++k) {
            int tt = t - 3 + k;
            if (tt >= 0)
                accv += xz[((size_t)(b * SEQ + tt)) * XZW + d] * cw[d * 4 + k];
        }
        float s = accv / (1.f + __expf(-accv));
        xa[(size_t)token * INNER + d]  = s;
        xab[(size_t)token * INNER + d] = (__bf16)s;
    }
}

// ---------------- dt mean over d ----------------
__global__ __launch_bounds__(256) void k_row_mean(const float* __restrict__ dt,
                                                  float* __restrict__ dtm) {
    __shared__ float red[256];
    const int t = blockIdx.x;
    float s = 0.f;
    for (int d = threadIdx.x; d < INNER; d += 256) s += dt[(size_t)t * INNER + d];
    red[threadIdx.x] = s; __syncthreads();
    for (int o = 128; o > 0; o >>= 1) {
        if (threadIdx.x < o) red[threadIdx.x] += red[threadIdx.x + o];
        __syncthreads();
    }
    if (threadIdx.x == 0) dtm[t] = red[0] * (1.f / INNER);
}

// ---------------- fused SSM scan + z-gating -> bf16 ----------------
// One thread per (b,d). a_bar[n] = exp(-dt)^(n+1): 1 exp + 15 muls.
// Bm/Cm/dt_mean staged in LDS for 64 timesteps per barrier.
__global__ __launch_bounds__(256) void k_ssm_scan(const float* __restrict__ dt,
                                                  const float* __restrict__ xa,
                                                  const float* __restrict__ bcv,
                                                  const float* __restrict__ dtm,
                                                  const float* __restrict__ xz,
                                                  __bf16* __restrict__ gated) {
    __shared__ float sBC[64][33];   // [t][0..15]=Bm, [16..31]=Cm (pad to 33)
    __shared__ float sDT[64];
    const int b = blockIdx.y;
    const int d = blockIdx.x * 256 + threadIdx.x;

    float h[NST];
#pragma unroll
    for (int n = 0; n < NST; ++n) h[n] = 0.f;

    for (int t0 = 0; t0 < SEQ; t0 += 64) {
        __syncthreads();
        for (int i = threadIdx.x; i < 64 * 32; i += 256) {
            int tt = i >> 5, j = i & 31;
            sBC[tt][j] = bcv[((size_t)(b * SEQ + t0 + tt)) * 32 + j];
        }
        for (int i = threadIdx.x; i < 64; i += 256)
            sDT[i] = dtm[b * SEQ + t0 + i];
        __syncthreads();

        for (int tt = 0; tt < 64; ++tt) {
            const int tok = b * SEQ + t0 + tt;
            const float dv = dt[(size_t)tok * INNER + d];
            const float xv = xa[(size_t)tok * INNER + d];
            const float zv = xz[(size_t)tok * XZW + INNER + d];
            const float e  = __expf(-dv);
            const float sc = sDT[tt] * xv;
            float a = 1.f, y = 0.f;
#pragma unroll
            for (int n = 0; n < NST; ++n) {
                a *= e;                                  // e^(n+1)
                h[n] = a * h[n] + sc * sBC[tt][n];       // h = a_bar*h + bx
                y += h[n] * sBC[tt][16 + n];             // dot with Cm
            }
            const float sz = zv / (1.f + __expf(-zv));   // silu(z)
            gated[(size_t)tok * INNER + d] = (__bf16)(y * sz);
        }
    }
}

// ---------------- host launch ----------------
extern "C" void kernel_launch(void* const* d_in, const int* in_sizes, int n_in,
                              void* d_out, int out_size, void* d_ws, size_t ws_size,
                              hipStream_t stream) {
    const float* x      = (const float*)d_in[0];
    const float* norm_w = (const float*)d_in[1];
    const float* norm_b = (const float*)d_in[2];
    const float* in_w   = (const float*)d_in[3];
    const float* in_b   = (const float*)d_in[4];
    const float* conv_w = (const float*)d_in[5];
    const float* conv_b = (const float*)d_in[6];
    const float* bc_w   = (const float*)d_in[7];
    const float* bc_b   = (const float*)d_in[8];
    const float* dtr_w  = (const float*)d_in[9];
    const float* dtr_b  = (const float*)d_in[10];
    const float* dt_w   = (const float*)d_in[11];
    const float* dt_b   = (const float*)d_in[12];
    const float* out_w  = (const float*)d_in[13];
    const float* out_b  = (const float*)d_in[14];
    float* outp = (float*)d_out;

    // workspace carve-out (256B aligned)
    char* ws = (char*)d_ws;
    size_t off = 0;
    auto alloc = [&](size_t bytes) -> void* {
        off = (off + 255) & ~(size_t)255;
        void* p = ws + off;
        off += bytes;
        return p;
    };
    __bf16* normed_bf  = (__bf16*)alloc((size_t)TOK * HDIM * 2);
    __bf16* in_w_bf    = (__bf16*)alloc((size_t)HDIM * XZW * 2);
    __bf16* bc_w_bf    = (__bf16*)alloc((size_t)INNER * 32 * 2);
    __bf16* dtr_w_bf   = (__bf16*)alloc((size_t)INNER * DTRD * 2);
    __bf16* dt_w_bf    = (__bf16*)alloc((size_t)DTRD * INNER * 2);
    __bf16* out_w_bf   = (__bf16*)alloc((size_t)INNER * HDIM * 2);
    float*  xz         = (float*) alloc((size_t)TOK * XZW * 4);
    float*  xa         = (float*) alloc((size_t)TOK * INNER * 4);
    __bf16* xa_bf      = (__bf16*)alloc((size_t)TOK * INNER * 2);
    float*  bc         = (float*) alloc((size_t)TOK * 32 * 4);
    float*  dtrv       = (float*) alloc((size_t)TOK * DTRD * 4);
    __bf16* dtr_bf     = (__bf16*)alloc((size_t)TOK * DTRD * 2);
    float*  dtv        = (float*) alloc((size_t)TOK * INNER * 4);
    float*  dtm        = (float*) alloc((size_t)TOK * 4);
    __bf16* gated_bf   = (__bf16*)alloc((size_t)TOK * INNER * 2);
    (void)ws_size; (void)n_in; (void)in_sizes; (void)out_size;

    auto cvt = [&](const float* src, __bf16* dst, int n) {
        k_f32_to_bf16<<<(n + 255) / 256, 256, 0, stream>>>(src, dst, n);
    };
    // weight conversions
    cvt(in_w,  in_w_bf,  HDIM * XZW);
    cvt(bc_w,  bc_w_bf,  INNER * 32);
    cvt(dtr_w, dtr_w_bf, INNER * DTRD);
    cvt(dt_w,  dt_w_bf,  DTRD * INNER);
    cvt(out_w, out_w_bf, INNER * HDIM);

    // 1) layernorm -> bf16
    k_layernorm_bf16<<<TOK, 256, 0, stream>>>(x, norm_w, norm_b, normed_bf);

    // 2) xz = normed @ in_w + in_b   (2048 x 1024 x 4096)
    k_wmma_gemm<<<dim3((XZW + 127) / 128, TOK / 64), 256, 0, stream>>>(
        normed_bf, in_w_bf, in_b, xz, TOK, XZW, HDIM, 0);

    // 3) causal conv + silu -> xa (f32 + bf16)
    k_conv_silu<<<TOK, 256, 0, stream>>>(xz, conv_w, conv_b, xa, xa_bf);

    // 4) bc = xa @ bc_w + bc_b   (2048 x 2048 x 32)
    k_wmma_gemm<<<dim3(1, TOK / 64), 256, 0, stream>>>(
        xa_bf, bc_w_bf, bc_b, bc, TOK, 32, INNER, 0);

    // 5) dt chain: (xa @ dtr_w + dtr_b) @ dt_w + dt_b -> softplus -> clip
    k_wmma_gemm<<<dim3(1, TOK / 64), 256, 0, stream>>>(
        xa_bf, dtr_w_bf, dtr_b, dtrv, TOK, DTRD, INNER, 0);
    cvt(dtrv, dtr_bf, TOK * DTRD);
    k_wmma_gemm<<<dim3((INNER + 127) / 128, TOK / 64), 256, 0, stream>>>(
        dtr_bf, dt_w_bf, dt_b, dtv, TOK, INNER, DTRD, 1);

    // 6) dt mean over channel dim
    k_row_mean<<<TOK, 256, 0, stream>>>(dtv, dtm);

    // 7) fused SSM scan + gating -> bf16
    k_ssm_scan<<<dim3(INNER / 256, BATCH), 256, 0, stream>>>(
        dtv, xa, bc, dtm, xz, gated_bf);

    // 8) out = gated @ out_w + out_b   (2048 x 2048 x 1024) -> d_out (f32)
    k_wmma_gemm<<<dim3((HDIM + 127) / 128, TOK / 64), 256, 0, stream>>>(
        gated_bf, out_w_bf, out_b, outp, TOK, HDIM, INNER, 0);
}
